// BestLstmModel_38448547233812
// MI455X (gfx1250) — compile-verified, matrix operand order checked by reading, not run on hardware
//
#include <hip/hip_runtime.h>
#include <hip/hip_bf16.h>

typedef _Float16 half_t;
typedef __attribute__((ext_vector_type(16))) _Float16 v16h;
typedef __attribute__((ext_vector_type(8)))  _Float16 v8h;
typedef __attribute__((ext_vector_type(8)))  float    v8f;
typedef __attribute__((ext_vector_type(4)))  float    f4;

#define EPS_BN 1e-3f
#define LOG2E  1.44269504088896340736f

// Fast gate activations on hardware transcendentals (v_exp_f32 / v_rcp_f32).
// Saturation at large |x| is exact: exp2->inf -> rcp->0.
__device__ __forceinline__ float fast_exp(float x) {
    return __builtin_amdgcn_exp2f(x * LOG2E);
}
__device__ __forceinline__ float fast_sig(float x) {
    return __builtin_amdgcn_rcpf(1.0f + fast_exp(-x));
}
__device__ __forceinline__ float fast_tanh(float x) {
    return 1.0f - 2.0f * __builtin_amdgcn_rcpf(fast_exp(2.0f * x) + 1.0f);
}

__device__ __forceinline__ v8f wmma_f16(v16h a, v16h b, v8f c) {
    return __builtin_amdgcn_wmma_f32_16x16x32_f16(false, a, false, b, (short)0, c, false, false);
}

// Load a 32-byte WMMA operand fragment from LDS as two 16B vector loads.
__device__ __forceinline__ v16h ld_frag(const half_t* p) {
    v8h lo = *(const v8h*)(p);
    v8h hi = *(const v8h*)(p + 8);
    return __builtin_shufflevector(lo, hi, 0, 1, 2, 3, 4, 5, 6, 7,
                                   8, 9, 10, 11, 12, 13, 14, 15);
}

__device__ __forceinline__ v8f ld_acc(const float* p) {
    f4 lo = *(const f4*)(p);
    f4 hi = *(const f4*)(p + 4);
    return __builtin_shufflevector(lo, hi, 0, 1, 2, 3, 4, 5, 6, 7);
}

__device__ __forceinline__ void st_acc(float* p, v8f c) {
    *(f4*)(p)     = __builtin_shufflevector(c, c, 0, 1, 2, 3);
    *(f4*)(p + 4) = __builtin_shufflevector(c, c, 4, 5, 6, 7);
}

// ---------------------------------------------------------------------------
// Input-projection GEMM. M-tile = 16 batch rows at one timestep t.
// X:  (B, T, K) f32 row-major (standard activation layout).
// Out: fragment-swizzled XW, float, laid out as
//      [bblk][t][ntile][lane][8]  (lane's WMMA C-fragment contiguous).
// reverse=1 -> read X at time T-1-t (backward direction).
// Block: 256 threads = 8 waves; wave w handles t = blockIdx.y*8 + w.
// ---------------------------------------------------------------------------
template <int K, int N4>
__global__ __launch_bounds__(256) void gemm_xw_kernel(
    const float* __restrict__ X, const float* __restrict__ Wk,
    const float* __restrict__ bias, float* __restrict__ Out,
    int T, int reverse)
{
    constexpr int KSTEPS = K / 32;
    constexpr int NT = N4 / 16;
    __shared__ __attribute__((aligned(32))) half_t wlds[NT * KSTEPS * 32 * 16];

    // Stage weights pre-swizzled into exact B-fragment order:
    // [frag = nt*KSTEPS+ks][lane][e], 16 halves contiguous per lane.
    for (int idx = threadIdx.x; idx < NT * KSTEPS * 32 * 16; idx += 256) {
        int e      = idx & 15;
        int lane_l = (idx >> 4) & 31;
        int frag   = idx >> 9;
        int ks     = frag % KSTEPS;
        int nt     = frag / KSTEPS;
        int n   = nt * 16 + (lane_l & 15);
        int k0l = (lane_l >> 4) * 8;
        int k   = ks * 32 + k0l + (e >> 3) * 16 + (e & 7);
        wlds[idx] = (half_t)Wk[k * N4 + n];
    }
    __syncthreads();

    const int lane = threadIdx.x & 31;
    const int wave = threadIdx.x >> 5;
    const int t    = blockIdx.y * 8 + wave;
    const int tsrc = reverse ? (T - 1 - t) : t;
    const long b0  = (long)blockIdx.x * 16;

    const int mrow = lane & 15;
    const int k0   = (lane >> 4) * 8;
    const int ncol = lane & 15;

    // Preload per-lane bias values (clause-grouped loads, no per-tile stall)
    float bv[NT];
#pragma unroll
    for (int nt = 0; nt < NT; ++nt) bv[nt] = bias[nt * 16 + ncol];

    const float* px = X + ((b0 + mrow) * (long)T + tsrc) * K;

    // A fragments (16x32 f16 per K-step): vector loads + packed converts
    v16h a[KSTEPS];
#pragma unroll
    for (int ks = 0; ks < KSTEPS; ++ks) {
        const int kb = ks * 32 + k0;
        f4 x0 = *(const f4*)(px + kb);
        f4 x1 = *(const f4*)(px + kb + 4);
        f4 x2 = *(const f4*)(px + kb + 16);
        f4 x3 = *(const f4*)(px + kb + 20);
#pragma unroll
        for (int j = 0; j < 4; ++j) {
            a[ks][j]      = (half_t)x0[j];
            a[ks][4 + j]  = (half_t)x1[j];
            a[ks][8 + j]  = (half_t)x2[j];
            a[ks][12 + j] = (half_t)x3[j];
        }
    }

    // Out base for this wave's (bblk, t): floats
    float* po = Out + (((long)blockIdx.x * T + t) * NT) * 32 * 8;

#pragma unroll
    for (int nt = 0; nt < NT; ++nt) {
        const float bval = bv[nt];
        v8f c = {bval, bval, bval, bval, bval, bval, bval, bval};
#pragma unroll
        for (int ks = 0; ks < KSTEPS; ++ks) {
            v16h bf = ld_frag(&wlds[((nt * KSTEPS + ks) * 32 + lane) * 16]);
            c = wmma_f16(a[ks], bf, c);
        }
        st_acc(po + ((long)nt * 32 + lane) * 8, c);
    }
}

// ---------------------------------------------------------------------------
// LSTM recurrence, one direction. 1 wave per block, 16 batch rows per block.
// XW: fragment-swizzled [bblk][t][ntile][lane][8] from gemm_xw_kernel.
// Wr: (H, 4H) f32 in global, staged pre-swizzled as f16 B-fragments in LDS.
// Output store fuses BatchNorm affine; backward stores at reversed time;
// lastOnly stores only the final h (layer 3).
// ---------------------------------------------------------------------------
template <int H>
__global__ __launch_bounds__(32) void lstm_rec_kernel(
    const float* __restrict__ XW, const float* __restrict__ Wr,
    float* __restrict__ Out, int T, int outC, int chOff,
    int backward, int lastOnly,
    const float* __restrict__ bn_g, const float* __restrict__ bn_b,
    const float* __restrict__ bn_m, const float* __restrict__ bn_v)
{
    constexpr int N4 = 4 * H;
    constexpr int HT = H / 16;              // tiles per gate
    constexpr int KP = (H < 32) ? 32 : H;   // zero-padded K for WMMA
    constexpr int KSTEPS = KP / 32;
    constexpr int NT = N4 / 16;

    __shared__ __attribute__((aligned(32))) half_t wr_lds[NT * KSTEPS * 32 * 16];
    __shared__ __attribute__((aligned(32))) half_t h_lds[16 * KP];

    const int lane = threadIdx.x & 31;

    // Pre-swizzled recurrent weights (zero-pad K rows >= H for layer 3)
    for (int idx = lane; idx < NT * KSTEPS * 32 * 16; idx += 32) {
        int e      = idx & 15;
        int lane_l = (idx >> 4) & 31;
        int frag   = idx >> 9;
        int ks     = frag % KSTEPS;
        int nt     = frag / KSTEPS;
        int n   = nt * 16 + (lane_l & 15);
        int k0l = (lane_l >> 4) * 8;
        int k   = ks * 32 + k0l + (e >> 3) * 16 + (e & 7);
        wr_lds[idx] = (k < H) ? (half_t)Wr[k * N4 + n] : (half_t)0.0f;
    }
    for (int i = lane; i < 16 * KP; i += 32)
        h_lds[i] = (half_t)0.0f;
    __syncthreads();

    const int mrow = lane & 15;
    const int k0   = (lane >> 4) * 8;
    const int ncol = lane & 15;
    const int mtop = (lane >> 4) ? 8 : 0;
    const long b0  = (long)blockIdx.x * 16;

    // BN affine per lane-channel
    float scale[HT], shift[HT];
#pragma unroll
    for (int j = 0; j < HT; ++j) {
        int ch = chOff + j * 16 + ncol;
        float s = bn_g[ch] * rsqrtf(bn_v[ch] + EPS_BN);
        scale[j] = s;
        shift[j] = bn_b[ch] - bn_m[ch] * s;
    }

    v8f cstate[HT];
#pragma unroll
    for (int j = 0; j < HT; ++j) cstate[j] = (v8f){0, 0, 0, 0, 0, 0, 0, 0};

    const float* xw_blk = XW + ((long)blockIdx.x * T) * NT * 32 * 8;

    for (int t = 0; t < T; ++t) {
        // A fragments of h from LDS (row-contiguous -> 16B vector loads)
        v16h a[KSTEPS];
#pragma unroll
        for (int ks = 0; ks < KSTEPS; ++ks) {
            const int kb = ks * 32 + k0;
            v8h lo = *(const v8h*)(&h_lds[mrow * KP + kb]);
            v8h hi = *(const v8h*)(&h_lds[mrow * KP + kb + 16]);
            a[ks] = __builtin_shufflevector(lo, hi, 0, 1, 2, 3, 4, 5, 6, 7,
                                            8, 9, 10, 11, 12, 13, 14, 15);
        }

        const float* xw_t = xw_blk + (long)t * NT * 32 * 8;
        if (t + 1 < T)  // prefetch next timestep's gate pre-activations
            __builtin_prefetch(xw_t + NT * 32 * 8 + lane * 8, 0, 1);

#pragma unroll
        for (int j = 0; j < HT; ++j) {      // gate group j of i,f,g,o
            v8f zt[4];
#pragma unroll
            for (int g = 0; g < 4; ++g) {
                const int nt = g * HT + j;
                v8f acc = ld_acc(xw_t + ((long)nt * 32 + lane) * 8);
#pragma unroll
                for (int ks = 0; ks < KSTEPS; ++ks) {
                    v16h bf = ld_frag(&wr_lds[((nt * KSTEPS + ks) * 32 + lane) * 16]);
                    acc = wmma_f16(a[ks], bf, acc);
                }
                zt[g] = acc;
            }
            // elementwise gates on hardware transcendentals
            v8f hn;
#pragma unroll
            for (int r = 0; r < 8; ++r) {
                float ig = fast_sig(zt[0][r]);
                float fg = fast_sig(zt[1][r]);
                float gg = fast_tanh(zt[2][r]);
                float og = fast_sig(zt[3][r]);
                float cv = fg * cstate[j][r] + ig * gg;
                cstate[j][r] = cv;
                hn[r] = og * fast_tanh(cv);
            }
            // stash h (f16) for next step's A fragments
#pragma unroll
            for (int r = 0; r < 8; ++r)
                h_lds[(mtop + r) * KP + j * 16 + ncol] = (half_t)hn[r];

            // BN-fused output store (standard activation layout for next gemm)
            if (!lastOnly || (t == T - 1)) {
                const int tstore = lastOnly ? 0 : (backward ? (T - 1 - t) : t);
                const int ch = chOff + j * 16 + ncol;
#pragma unroll
                for (int r = 0; r < 8; ++r) {
                    long row = b0 + mtop + r;
                    long idx = lastOnly ? row * outC + ch
                                        : (row * (long)T + tstore) * outC + ch;
                    Out[idx] = scale[j] * hn[r] + shift[j];
                }
            }
        }
        __syncthreads();
    }
}

// ---------------------------------------------------------------------------
// Head: E(2048,32) [BN3 applied] -> dense(32,16)+clip(0,6) -> BN4 ->
// dense(16,2) -> softmax
// ---------------------------------------------------------------------------
__global__ __launch_bounds__(256) void head_kernel(
    const float* __restrict__ E,
    const float* __restrict__ d4w, const float* __restrict__ d4b,
    const float* __restrict__ bn4g, const float* __restrict__ bn4b,
    const float* __restrict__ bn4m, const float* __restrict__ bn4v,
    const float* __restrict__ dcw, const float* __restrict__ dcb,
    float* __restrict__ out, int B)
{
    int i = blockIdx.x * blockDim.x + threadIdx.x;
    if (i >= B) return;
    const float* e = E + (long)i * 32;
    float h1[16];
#pragma unroll
    for (int n = 0; n < 16; ++n) {
        float acc = d4b[n];
        for (int k = 0; k < 32; ++k) acc += e[k] * d4w[k * 16 + n];
        acc = fminf(fmaxf(acc, 0.0f), 6.0f);
        h1[n] = bn4g[n] * (acc - bn4m[n]) * rsqrtf(bn4v[n] + EPS_BN) + bn4b[n];
    }
    float l0 = dcb[0], l1 = dcb[1];
#pragma unroll
    for (int k = 0; k < 16; ++k) {
        l0 += h1[k] * dcw[k * 2 + 0];
        l1 += h1[k] * dcw[k * 2 + 1];
    }
    float mx = fmaxf(l0, l1);
    float e0 = fast_exp(l0 - mx), e1 = fast_exp(l1 - mx);
    float inv = 1.0f / (e0 + e1);
    out[(long)i * 2 + 0] = e0 * inv;
    out[(long)i * 2 + 1] = e1 * inv;
}

// ---------------------------------------------------------------------------
extern "C" void kernel_launch(void* const* d_in, const int* in_sizes, int n_in,
                              void* d_out, int out_size, void* d_ws, size_t ws_size,
                              hipStream_t stream) {
    const int B = 2048, T = 128;

    const float* x     = (const float*)d_in[0];
    const float* w1f_k = (const float*)d_in[1];
    const float* w1f_r = (const float*)d_in[2];
    const float* w1f_b = (const float*)d_in[3];
    const float* w1b_k = (const float*)d_in[4];
    const float* w1b_r = (const float*)d_in[5];
    const float* w1b_b = (const float*)d_in[6];
    const float* w2f_k = (const float*)d_in[7];
    const float* w2f_r = (const float*)d_in[8];
    const float* w2f_b = (const float*)d_in[9];
    const float* w2b_k = (const float*)d_in[10];
    const float* w2b_r = (const float*)d_in[11];
    const float* w2b_b = (const float*)d_in[12];
    const float* w3f_k = (const float*)d_in[13];
    const float* w3f_r = (const float*)d_in[14];
    const float* w3f_b = (const float*)d_in[15];
    const float* w3b_k = (const float*)d_in[16];
    const float* w3b_r = (const float*)d_in[17];
    const float* w3b_b = (const float*)d_in[18];
    const float* bn1_g = (const float*)d_in[19];
    const float* bn1_b = (const float*)d_in[20];
    const float* bn1_m = (const float*)d_in[21];
    const float* bn1_v = (const float*)d_in[22];
    const float* bn2_g = (const float*)d_in[23];
    const float* bn2_b = (const float*)d_in[24];
    const float* bn2_m = (const float*)d_in[25];
    const float* bn2_v = (const float*)d_in[26];
    const float* bn3_g = (const float*)d_in[27];
    const float* bn3_b = (const float*)d_in[28];
    const float* bn3_m = (const float*)d_in[29];
    const float* bn3_v = (const float*)d_in[30];
    const float* bn4_g = (const float*)d_in[31];
    const float* bn4_b = (const float*)d_in[32];
    const float* bn4_m = (const float*)d_in[33];
    const float* bn4_v = (const float*)d_in[34];
    const float* d4_w  = (const float*)d_in[35];
    const float* d4_b  = (const float*)d_in[36];
    const float* dc_w  = (const float*)d_in[37];
    const float* dc_b  = (const float*)d_in[38];

    char* ws = (char*)d_ws;
    size_t off = 0;
    float* bufA = (float*)(ws + off); off += (size_t)B * T * 256 * 4;  // 268 MB
    float* bufB = (float*)(ws + off); off += (size_t)B * T * 256 * 4;  // 268 MB
    float* bufC = (float*)(ws + off); off += (size_t)B * T * 128 * 4;  // 134 MB
    float* bufD = (float*)(ws + off); off += (size_t)B * T * 64 * 4;   //  67 MB
    float* bufE = (float*)(ws + off); off += (size_t)B * 32 * 4;       // 256 KB
    (void)ws_size;

    const dim3 gGemm(B / 16, T / 8);  // (128, 16) blocks, 256 thr
    const int  gRec = B / 16;         // 128 blocks, 32 thr

    // ---- Layer 1: BiLSTM(64), seq out, BN1 fused into rec store ----
    gemm_xw_kernel<64, 256><<<gGemm, 256, 0, stream>>>(x, w1f_k, w1f_b, bufA, T, 0);
    gemm_xw_kernel<64, 256><<<gGemm, 256, 0, stream>>>(x, w1b_k, w1b_b, bufB, T, 1);
    lstm_rec_kernel<64><<<gRec, 32, 0, stream>>>(bufA, w1f_r, bufC, T, 128, 0, 0, 0,
                                                 bn1_g, bn1_b, bn1_m, bn1_v);
    lstm_rec_kernel<64><<<gRec, 32, 0, stream>>>(bufB, w1b_r, bufC, T, 128, 64, 1, 0,
                                                 bn1_g, bn1_b, bn1_m, bn1_v);

    // ---- Layer 2: BiLSTM(32), seq out, BN2 fused ----
    gemm_xw_kernel<128, 128><<<gGemm, 256, 0, stream>>>(bufC, w2f_k, w2f_b, bufA, T, 0);
    gemm_xw_kernel<128, 128><<<gGemm, 256, 0, stream>>>(bufC, w2b_k, w2b_b, bufB, T, 1);
    lstm_rec_kernel<32><<<gRec, 32, 0, stream>>>(bufA, w2f_r, bufD, T, 64, 0, 0, 0,
                                                 bn2_g, bn2_b, bn2_m, bn2_v);
    lstm_rec_kernel<32><<<gRec, 32, 0, stream>>>(bufB, w2b_r, bufD, T, 64, 32, 1, 0,
                                                 bn2_g, bn2_b, bn2_m, bn2_v);

    // ---- Layer 3: BiLSTM(16), last-only, BN3 fused ----
    gemm_xw_kernel<64, 64><<<gGemm, 256, 0, stream>>>(bufD, w3f_k, w3f_b, bufA, T, 0);
    gemm_xw_kernel<64, 64><<<gGemm, 256, 0, stream>>>(bufD, w3b_k, w3b_b, bufB, T, 1);
    lstm_rec_kernel<16><<<gRec, 32, 0, stream>>>(bufA, w3f_r, bufE, T, 32, 0, 0, 1,
                                                 bn3_g, bn3_b, bn3_m, bn3_v);
    lstm_rec_kernel<16><<<gRec, 32, 0, stream>>>(bufB, w3b_r, bufE, T, 32, 16, 1, 1,
                                                 bn3_g, bn3_b, bn3_m, bn3_v);

    // ---- Head ----
    head_kernel<<<(B + 255) / 256, 256, 0, stream>>>(
        bufE, d4_w, d4_b, bn4_g, bn4_b, bn4_m, bn4_v, dc_w, dc_b,
        (float*)d_out, B);
}
